// Attention_69509750719031
// MI455X (gfx1250) — matrix operand order repaired and checked by reading, not compiled, not run on hardware
//
#include <hip/hip_runtime.h>
#include <hip/hip_bf16.h>

// ---------------------------------------------------------------------------
// Attention block (GroupNorm -> QKV -> MHSA -> proj + residual) for gfx1250.
// All GEMMs use v_wmma_f32_16x16x32_bf16 (bf16 in, f32 accumulate).
// b=16, c=512, H=8, d=64, N=1024 (32x32), groups=8.
//
// Layout strategy: every WMMA B-operand is stored K-fastest so each lane's
// fragment is 2 x 8 contiguous bf16, 16 k apart (2 x global_load_b128); every A-operand is
// row-major so each lane's fragment is 2 x 16B contiguous runs.
// ---------------------------------------------------------------------------

typedef __attribute__((ext_vector_type(16))) __bf16 v16bf;
typedef __attribute__((ext_vector_type(8)))  __bf16 v8bf;
typedef __attribute__((ext_vector_type(8)))  float  v8f;

#define DEV __device__ __forceinline__

static constexpr int B  = 16;
static constexpr int C  = 512;
static constexpr int N  = 1024;   // 32*32 tokens
static constexpr int H  = 8;
static constexpr int D  = 64;
static constexpr int G  = 8;      // groupnorm groups
static constexpr int GSZ = (C / G) * N;   // 65536 elements per group

// ---------------- WMMA fragment helpers (per CDNA5 ISA 7.12.2) -------------
// A (16x32 bf16) from row-major storage, leading dim ld (elements):
//   lane&15 = row M; kh=(lane>>4): VGPRs 0-3 hold k=8kh..8kh+7,
//   VGPRs 4-7 hold k=16+8kh..16+8kh+7  -> two contiguous 16B runs per lane.
DEV v16bf load_a_frag(const __bf16* p, int ld, int lane) {
    int m  = lane & 15;
    int kh = lane >> 4;
    const __bf16* row = p + m * ld;
    v8bf lo = *(const v8bf*)(row + 8 * kh);
    v8bf hi = *(const v8bf*)(row + 16 + 8 * kh);
    v16bf a;
#pragma unroll
    for (int i = 0; i < 8; ++i) { a[i] = lo[i]; a[8 + i] = hi[i]; }
    return a;
}

// B (32x16 bf16) from K-fastest ("transposed") storage: element B[k][n] at
// p[n*ld + k]. lane&15 = col N; kh=(lane>>4): elems 0-7 hold k=8kh..8kh+7,
// elems 8-15 hold k=16+8kh..16+8kh+7 (2 x b128).
DEV v16bf load_b_fragT(const __bf16* p, int ld, int lane) {
    int n  = lane & 15;
    int kb = (lane >> 4) * 8;
    const __bf16* col = p + n * ld + kb;
    v8bf lo = *(const v8bf*)(col);
    v8bf hi = *(const v8bf*)(col + 16);
    v16bf b;
#pragma unroll
    for (int i = 0; i < 8; ++i) { b[i] = lo[i]; b[8 + i] = hi[i]; }
    return b;
}

DEV v8f wmma_bf16(v16bf a, v16bf b, v8f c) {
    return __builtin_amdgcn_wmma_f32_16x16x32_bf16(
        /*neg_a=*/false, a, /*neg_b=*/false, b,
        /*c_mod=*/(short)0, c, /*reuse_a=*/false, /*reuse_b=*/false);
}

// ---------------- K1: GroupNorm -> bf16 activations, token-major ------------
// Writes xnT layout (b, N, C) via LDS tile transpose (64c x 64n tiles).
__global__ void groupnorm_kernel(const float* __restrict__ x,
                                 const float* __restrict__ gn_w,
                                 const float* __restrict__ gn_b,
                                 __bf16* __restrict__ xnT) {
    int bg = blockIdx.x;
    int b  = bg >> 3;
    int g  = bg & 7;
    const float* base = x + (size_t)(b * C + g * (C / G)) * N;

    // ---- pass 1: mean / var ------------------------------------------------
    float s = 0.f, ss = 0.f;
    for (int i = threadIdx.x; i < GSZ; i += 256) {
        float v = base[i];
        s  += v;
        ss += v * v;
    }
#pragma unroll
    for (int off = 16; off > 0; off >>= 1) {
        s  += __shfl_xor(s,  off, 32);
        ss += __shfl_xor(ss, off, 32);
    }
    __shared__ float red[2][8];
    int wid = threadIdx.x >> 5;
    if ((threadIdx.x & 31) == 0) { red[0][wid] = s; red[1][wid] = ss; }
    __syncthreads();
    if (threadIdx.x == 0) {
        float ts = 0.f, tss = 0.f;
#pragma unroll
        for (int i = 0; i < 8; ++i) { ts += red[0][i]; tss += red[1][i]; }
        float mu  = ts / (float)GSZ;
        float var = tss / (float)GSZ - mu * mu;
        red[0][0] = mu;
        red[1][0] = rsqrtf(var + 1e-5f);
    }
    __syncthreads();
    float mu = red[0][0], rstd = red[1][0];

    // ---- pass 2: normalize + transpose through LDS -------------------------
    __shared__ __bf16 tile[64][72];   // 64c x 64n, padded
    for (int nt = 0; nt < 16; ++nt) {
        for (int e = threadIdx.x; e < 4096; e += 256) {
            int cl = e >> 6, nn = e & 63;          // coalesced read along n
            float v = (base[cl * N + nt * 64 + nn] - mu) * rstd
                      * gn_w[g * 64 + cl] + gn_b[g * 64 + cl];
            tile[cl][nn] = (__bf16)v;
        }
        __syncthreads();
        for (int e = threadIdx.x; e < 4096; e += 256) {
            int nn = e >> 6, cl = e & 63;          // coalesced write along c
            xnT[((size_t)b * N + nt * 64 + nn) * C + g * 64 + cl] = tile[cl][nn];
        }
        __syncthreads();
    }
}

// ---------------- K2: f32 -> bf16 weight conversion -------------------------
__global__ void cvt_bf16_kernel(const float* __restrict__ src,
                                __bf16* __restrict__ dst, int n) {
    int i = blockIdx.x * 256 + threadIdx.x;
    if (i < n) dst[i] = (__bf16)src[i];
}

// ---------------- K3: QKV GEMM (1536 x N, K=512) ----------------------------
// One 16x16 tile per wave. q,k -> (b,H,N,d); v -> transposed (b,H,d,N).
__global__ void qkv_gemm_kernel(const __bf16* __restrict__ W,   // (1536,512)
                                const float*  __restrict__ bias,
                                const __bf16* __restrict__ xnT, // (b,N,C)
                                __bf16* __restrict__ q,
                                __bf16* __restrict__ k,
                                __bf16* __restrict__ vT) {
    int b    = blockIdx.z;
    int mt   = blockIdx.y;           // 0..95  (row tile of 3c)
    int ng   = blockIdx.x;           // 0..7
    int wave = threadIdx.x >> 5;
    int lane = threadIdx.x & 31;
    int nt   = ng * 8 + wave;        // 0..63 (col tile of N)

    const __bf16* Ab = W + (size_t)mt * 16 * C;
    const __bf16* Bb = xnT + ((size_t)b * N + nt * 16) * C;

    v8f acc = {};
    for (int k0 = 0; k0 < C; k0 += 32) {
        if (k0 + 32 < C) {
            __builtin_prefetch((const void*)(Ab + k0 + 32), 0, 1);
            __builtin_prefetch((const void*)(Bb + k0 + 32), 0, 1);
        }
        v16bf af = load_a_frag(Ab + k0, C, lane);
        v16bf bf = load_b_fragT(Bb + k0, C, lane);
        acc = wmma_bf16(af, bf, acc);
    }

    int col   = lane & 15;
    int roff  = (lane >> 4) * 8;
    int token = nt * 16 + col;
    int obase = mt * 16 + roff;          // uniform per half-lane-group, +r
    int which = (mt * 16) >> 9;          // 0=q 1=k 2=v (uniform over tile)
    int head  = ((mt * 16) & 511) >> 6;  // uniform over tile
    int dd0   = obase & 63;

    if (which == 2) {
        // V transposed: (b,H,d,N), scattered b16 stores (32B runs per row)
#pragma unroll
        for (int r = 0; r < 8; ++r) {
            float val = acc[r] + bias[obase + r];
            vT[(((size_t)b * H + head) * D + dd0 + r) * N + token] = (__bf16)val;
        }
    } else {
        // Q/K: (b,H,N,d) -> 8 consecutive dd per lane = one 16B store
        v8bf pk;
#pragma unroll
        for (int r = 0; r < 8; ++r) pk[r] = (__bf16)(acc[r] + bias[obase + r]);
        __bf16* dst = (which ? k : q) +
                      (((size_t)b * H + head) * N + token) * D + dd0;
        *(v8bf*)dst = pk;
    }
}

// ---------------- K4: flash attention (per b,h; 64-query blocks) ------------
// 4 waves x 16 query rows; online softmax in C-fragment layout.
__global__ void attn_kernel(const __bf16* __restrict__ q,   // (b,H,N,d)
                            const __bf16* __restrict__ k,   // (b,H,N,d)
                            const __bf16* __restrict__ vT,  // (b,H,d,N)
                            __bf16* __restrict__ aoutT) {   // (b,N,C)
    int bh   = blockIdx.y;           // b*H + h
    int b    = bh >> 3;
    int h    = bh & 7;
    int qb   = blockIdx.x;           // 0..15
    int wave = threadIdx.x >> 5;
    int lane = threadIdx.x & 31;
    int qrow0 = qb * 64 + wave * 16;

    const __bf16* Qb = q  + (size_t)bh * N * D;
    const __bf16* Kb = k  + (size_t)bh * N * D;
    const __bf16* Vb = vT + (size_t)bh * D * N;

    __shared__ __bf16 lds_p[4][16][80];   // 16x64 P tile per wave, 160B stride

    v8f o[4];
    float mrun[8], lrun[8];
#pragma unroll
    for (int t = 0; t < 4; ++t) o[t] = (v8f){};
#pragma unroll
    for (int r = 0; r < 8; ++r) { mrun[r] = -1e30f; lrun[r] = 0.f; }

    // Q A-fragments for this wave's 16 rows (K = d, 2 steps of 32)
    v16bf qa0 = load_a_frag(Qb + (size_t)qrow0 * D,      D, lane);
    v16bf qa1 = load_a_frag(Qb + (size_t)qrow0 * D + 32, D, lane);

    const float scale = 0.125f;   // d^-0.5, d=64

    for (int jb = 0; jb < 16; ++jb) {
        int j0 = jb * 64;

        // ---- S = Q K^T : 4 tiles of 16x16 -------------------------------
        // B[k][n] = K[(j0+t*16+n)*D + k] : K-fastest -> contiguous frags
        v8f s[4];
#pragma unroll
        for (int t = 0; t < 4; ++t) {
            v8f acc = {};
            const __bf16* kp = Kb + (size_t)(j0 + t * 16) * D;
            acc = wmma_bf16(qa0, load_b_fragT(kp,      D, lane), acc);
            acc = wmma_bf16(qa1, load_b_fragT(kp + 32, D, lane), acc);
            s[t] = acc;
        }

        // ---- online softmax (row = VGPR slot r + 8*(lane>>4)) -----------
        float mnew[8], alpha[8], rsum[8];
#pragma unroll
        for (int r = 0; r < 8; ++r) {
            float vm = fmaxf(fmaxf(s[0][r], s[1][r]), fmaxf(s[2][r], s[3][r])) * scale;
#pragma unroll
            for (int off = 1; off < 16; off <<= 1)
                vm = fmaxf(vm, __shfl_xor(vm, off, 32));
            mnew[r]  = fmaxf(mrun[r], vm);
            alpha[r] = __expf(mrun[r] - mnew[r]);
            rsum[r]  = 0.f;
        }
        int m_row = ((lane >> 4) << 3);
        int ccol  = lane & 15;
#pragma unroll
        for (int t = 0; t < 4; ++t) {
#pragma unroll
            for (int r = 0; r < 8; ++r) {
                float p = __expf(s[t][r] * scale - mnew[r]);
                rsum[r] += p;
                lds_p[wave][m_row + r][t * 16 + ccol] = (__bf16)p;
            }
        }
#pragma unroll
        for (int r = 0; r < 8; ++r) {
            float rs = rsum[r];
#pragma unroll
            for (int off = 1; off < 16; off <<= 1)
                rs += __shfl_xor(rs, off, 32);
            lrun[r] = lrun[r] * alpha[r] + rs;
            mrun[r] = mnew[r];
#pragma unroll
            for (int t = 0; t < 4; ++t) o[t][r] *= alpha[r];
        }
        __syncthreads();

        // ---- O += P V : P via LDS (A frags, 2x ds_load_b128 per frag) ---
        v16bf pa0 = load_a_frag(&lds_p[wave][0][0],  80, lane);
        v16bf pa1 = load_a_frag(&lds_p[wave][0][32], 80, lane);
#pragma unroll
        for (int t = 0; t < 4; ++t) {
            // B[k][n] = V^T[(t*16+n)*N + j0 + k] : K-fastest
            const __bf16* vp = Vb + (size_t)(t * 16) * N + j0;
            o[t] = wmma_bf16(pa0, load_b_fragT(vp,      N, lane), o[t]);
            o[t] = wmma_bf16(pa1, load_b_fragT(vp + 32, N, lane), o[t]);
        }
        __syncthreads();
    }

    // ---- epilogue: normalize, store token-major (b, token, h*64+dd) -----
    int ccol = lane & 15;
    int roff = (lane >> 4) << 3;
#pragma unroll
    for (int t = 0; t < 4; ++t) {
#pragma unroll
        for (int r = 0; r < 8; ++r) {
            float val = o[t][r] / lrun[r];
            int tok = qrow0 + roff + r;
            int dc  = h * D + t * 16 + ccol;
            aoutT[((size_t)b * N + tok) * C + dc] = (__bf16)val;
        }
    }
}

// ---------------- K5: proj GEMM + bias + residual ---------------------------
__global__ void proj_kernel(const __bf16* __restrict__ Wp,    // (512,512)
                            const float*  __restrict__ pb,
                            const __bf16* __restrict__ ainT,  // (b,N,C)
                            const float*  __restrict__ x,
                            float* __restrict__ out) {
    int b    = blockIdx.z;
    int mt   = blockIdx.y;           // 0..31
    int ng   = blockIdx.x;           // 0..7
    int wave = threadIdx.x >> 5;
    int lane = threadIdx.x & 31;
    int nt   = ng * 8 + wave;        // 0..63

    const __bf16* Ab = Wp + (size_t)mt * 16 * C;
    const __bf16* Bb = ainT + ((size_t)b * N + nt * 16) * C;

    v8f acc = {};
    for (int k0 = 0; k0 < C; k0 += 32) {
        if (k0 + 32 < C) {
            __builtin_prefetch((const void*)(Ab + k0 + 32), 0, 1);
            __builtin_prefetch((const void*)(Bb + k0 + 32), 0, 1);
        }
        v16bf af = load_a_frag(Ab + k0, C, lane);
        v16bf bf = load_b_fragT(Bb + k0, C, lane);
        acc = wmma_bf16(af, bf, acc);
    }

    int col = lane & 15;
    int ro  = (lane >> 4) * 8;
    int n   = nt * 16 + col;
#pragma unroll
    for (int r = 0; r < 8; ++r) {
        int oc = mt * 16 + ro + r;
        size_t idx = ((size_t)b * C + oc) * N + n;
        out[idx] = x[idx] + acc[r] + pb[oc];
    }
}

// ---------------------------------------------------------------------------
extern "C" void kernel_launch(void* const* d_in, const int* in_sizes, int n_in,
                              void* d_out, int out_size, void* d_ws, size_t ws_size,
                              hipStream_t stream) {
    const float* x      = (const float*)d_in[0];
    const float* gn_w   = (const float*)d_in[1];
    const float* gn_b   = (const float*)d_in[2];
    const float* qkv_w  = (const float*)d_in[3];
    const float* qkv_b  = (const float*)d_in[4];
    const float* proj_w = (const float*)d_in[5];
    const float* proj_b = (const float*)d_in[6];
    float* out = (float*)d_out;

    // workspace carve-up (bf16 elements)
    const size_t ACT = (size_t)B * C * N;          // 8,388,608
    char* ws = (char*)d_ws;
    __bf16* xnT  = (__bf16*)(ws);
    __bf16* qbuf = xnT  + ACT;
    __bf16* kbuf = qbuf + ACT;
    __bf16* vbuf = kbuf + ACT;                     // stored (b,H,d,N)
    __bf16* abuf = vbuf + ACT;                     // stored (b,N,C)
    __bf16* wq   = abuf + ACT;                     // 1536*512
    __bf16* wp   = wq   + (size_t)3 * C * C;       // 512*512

    // weight conversion (tiny, one pass)
    cvt_bf16_kernel<<<(3 * C * C + 255) / 256, 256, 0, stream>>>(qkv_w, wq, 3 * C * C);
    cvt_bf16_kernel<<<(C * C + 255) / 256, 256, 0, stream>>>(proj_w, wp, C * C);

    // GroupNorm (+ transpose to token-major bf16)
    groupnorm_kernel<<<B * G, 256, 0, stream>>>(x, gn_w, gn_b, xnT);

    // QKV GEMM: grid (ngroups, mtiles, batch)
    qkv_gemm_kernel<<<dim3(8, 96, B), 256, 0, stream>>>(wq, qkv_b, xnT, qbuf, kbuf, vbuf);

    // Flash attention: grid (qblocks, b*H)
    attn_kernel<<<dim3(16, B * H), 128, 0, stream>>>(qbuf, kbuf, vbuf, abuf);

    // Proj + residual
    proj_kernel<<<dim3(8, 32, B), 256, 0, stream>>>(wp, proj_b, abuf, x, out);
}
